// SparseMoE_27152783245408
// MI455X (gfx1250) — matrix operand order repaired and checked by reading, not compiled, not run on hardware
//
#include <hip/hip_runtime.h>
#include <hip/hip_bf16.h>

// Sparse MoE top-2 forward for MI455X (gfx1250, wave32, WMMA bf16).
// Roofline: ~5.5e11 FLOPs (top-2 sparse), ~320MB HBM traffic (~14us @23.3TB/s)
// -> matrix-unit bound -> v_wmma_f32_16x16x32_bf16 everywhere.
// x-row gather uses GLOBAL_LOAD_ASYNC_TO_LDS_B128 (ASYNCcnt) per CDNA5 ISA ch.8.

#define TOKENS 65536
#define DMODEL 512
#define NEXP   8
#define HDIM   2048
#define TILE_M 64
#define XPAD   8               // +8 bf16 -> row stride 520 elems = 260 dwords -> 4-bank shift/row
#define XSTR   (DMODEL + XPAD) // 520
#define HSTR   (64 + 8)        // 72 elems = 36 dwords -> conflict-free A-frag reads

typedef __attribute__((ext_vector_type(16))) __bf16 v16bf;
typedef __attribute__((ext_vector_type(8)))  float  v8f;
typedef __attribute__((ext_vector_type(4)))  unsigned int u32x4;

union Frag32B { v16bf v; u32x4 u[2]; };

__device__ __forceinline__ unsigned short f2bf(float f) {
  unsigned int u = __float_as_uint(f);
  u += 0x7FFFu + ((u >> 16) & 1u);   // round-to-nearest-even
  return (unsigned short)(u >> 16);
}

// A fragment (16x32 bf16, row-major src): lane l: M = m0 + (l&15),
// K runs [k0+8*(l>>4) .. +7] and [k0+16+8*(l>>4) .. +7]  (two 16B runs, 32B apart)
__device__ __forceinline__ v16bf load_a_frag(const unsigned short* p) {
  Frag32B f;
  f.u[0] = *(const u32x4*)(p);
  f.u[1] = *(const u32x4*)(p + 16);
  return f.v;
}
// B fragment (32x16 bf16, from N-major (transposed) weights): lane l: N = n0 + (l&15),
// K runs [k0+8*(l>>4) .. +7] and [k0+16+8*(l>>4) .. +7]  (two 16B runs, 32B apart)
__device__ __forceinline__ v16bf load_b_frag(const unsigned short* p) {
  Frag32B f;
  f.u[0] = *(const u32x4*)(p);
  f.u[1] = *(const u32x4*)(p + 16);
  return f.v;
}

__device__ __forceinline__ v8f wmma_bf16(v16bf a, v16bf b, v8f c) {
  return __builtin_amdgcn_wmma_f32_16x16x32_bf16(
      false, a, false, b, (short)0, c, false, false);
}

// ---------------- phase 0: precision/layout conversion ----------------

__global__ __launch_bounds__(256) void cvt_x_bf16(const float* __restrict__ x,
                                                  unsigned short* __restrict__ xb) {
  int i = (blockIdx.x * 256 + threadIdx.x) * 4;   // exact coverage: TOKENS*DMODEL
  float4 v = *(const float4*)(x + i);
  unsigned short o0 = f2bf(v.x), o1 = f2bf(v.y), o2 = f2bf(v.z), o3 = f2bf(v.w);
  unsigned int p0 = (unsigned int)o0 | ((unsigned int)o1 << 16);
  unsigned int p1 = (unsigned int)o2 | ((unsigned int)o3 << 16);
  *(uint2*)(xb + i) = make_uint2(p0, p1);
}

// src[z][R][C] f32 -> dst[z][C][R] bf16, LDS 32x32 tiled transpose
__global__ __launch_bounds__(256) void transpose_bf16(const float* __restrict__ src,
                                                      unsigned short* __restrict__ dst,
                                                      int R, int C) {
  __shared__ float tile[32][33];
  const float* s = src + (size_t)blockIdx.z * R * C;
  unsigned short* d = dst + (size_t)blockIdx.z * R * C;
  int tx = threadIdx.x & 31, ty = threadIdx.x >> 5;  // 32 x 8
  int r0 = blockIdx.y * 32, c0 = blockIdx.x * 32;
#pragma unroll
  for (int i = 0; i < 4; ++i)
    tile[ty + 8 * i][tx] = s[(size_t)(r0 + ty + 8 * i) * C + c0 + tx];
  __syncthreads();
#pragma unroll
  for (int i = 0; i < 4; ++i)
    d[(size_t)(c0 + ty + 8 * i) * R + r0 + tx] = f2bf(tile[tx][ty + 8 * i]);
}

// ---------------- phase 1: router (one wave32 per token) ----------------

__global__ __launch_bounds__(256) void router_topk(const float* __restrict__ x,
                                                   const float* __restrict__ rw,
                                                   const float* __restrict__ rb,
                                                   int* __restrict__ cnt,
                                                   int* __restrict__ idxlist,
                                                   float* __restrict__ wgtlist) {
  int tok  = (blockIdx.x * 256 + threadIdx.x) >> 5;
  int lane = threadIdx.x & 31;
  float acc[NEXP];
#pragma unroll
  for (int e = 0; e < NEXP; ++e) acc[e] = 0.f;
  const float* xr = x + (size_t)tok * DMODEL;
  for (int d = lane; d < DMODEL; d += 32) {
    float xv = xr[d];
#pragma unroll
    for (int e = 0; e < NEXP; ++e) acc[e] += xv * rw[d * NEXP + e];
  }
#pragma unroll
  for (int e = 0; e < NEXP; ++e) {
#pragma unroll
    for (int off = 16; off > 0; off >>= 1) acc[e] += __shfl_xor(acc[e], off, 32);
  }
  if (lane == 0) {
    float lg[NEXP];
#pragma unroll
    for (int e = 0; e < NEXP; ++e) lg[e] = acc[e] + rb[e];
    int i0 = 0;
#pragma unroll
    for (int e = 1; e < NEXP; ++e) if (lg[e] > lg[i0]) i0 = e;   // ties -> lowest idx
    int i1 = (i0 == 0) ? 1 : 0;
#pragma unroll
    for (int e = 0; e < NEXP; ++e) if (e != i1 && e != i0 && lg[e] > lg[i1]) i1 = e;
    // renormalized top-2 softmax == sigmoid of logit gap (Z cancels)
    float e1 = __expf(lg[i1] - lg[i0]);
    float p0 = 1.0f / (1.0f + e1);
    float p1 = e1 * p0;
    int s0 = atomicAdd(&cnt[i0], 1);
    idxlist[i0 * TOKENS + s0] = tok; wgtlist[i0 * TOKENS + s0] = p0;
    int s1 = atomicAdd(&cnt[i1], 1);
    idxlist[i1 * TOKENS + s1] = tok; wgtlist[i1 * TOKENS + s1] = p1;
  }
}

// ---------------- phase 2: fused expert FFN ----------------
// block = 256 threads (8 waves), TILE_M=64 gathered tokens of expert blockIdx.y.
// out[64x512] f32 accumulators live in registers across the whole H loop.
// h-tile double-buffered in LDS -> one barrier per H-chunk.

__global__ __launch_bounds__(256) void moe_ffn(const unsigned short* __restrict__ xb,
                                               const unsigned short* __restrict__ w1t, // [E][H][D]
                                               const unsigned short* __restrict__ w2t, // [E][D][H]
                                               const float* __restrict__ b1,           // [E][H]
                                               const float* __restrict__ b2,           // [E][D]
                                               const int* __restrict__ cnt,
                                               const int* __restrict__ idxlist,
                                               const float* __restrict__ wgtlist,
                                               float* __restrict__ out) {
  extern __shared__ char smem[];
  unsigned short* lds_x = (unsigned short*)smem;            // 64 x 520        (66560 B)
  unsigned short* lds_h = lds_x + TILE_M * XSTR;            // 2 x (64 x 72)   (18432 B)
  int*   lds_tok = (int*)(lds_h + 2 * TILE_M * HSTR);       // 64
  float* lds_wgt = (float*)(lds_tok + TILE_M);              // 64

  const int e    = blockIdx.y;
  const int base = blockIdx.x * TILE_M;
  const int n_e  = cnt[e];
  if (base >= n_e) return;

  const int tid  = threadIdx.x;
  const int lane = tid & 31;
  const int wave = tid >> 5;
  const int l16  = lane & 15;
  const int hh   = lane >> 4;

  if (tid < TILE_M) {
    int i = base + tid;
    int t; float w;
    if (i < n_e) { t = idxlist[e * TOKENS + i]; w = wgtlist[e * TOKENS + i]; }
    else         { t = idxlist[e * TOKENS + base]; w = 0.f; }   // padded row: weight 0
    lds_tok[tid] = t; lds_wgt[tid] = w;
  }
  __syncthreads();

  // gather 64 token rows (bf16) into LDS via CDNA5 async copies (ASYNCcnt),
  // 16B per lane per issue, coalesced within each row, no VGPR round-trip
  for (int it = tid; it < TILE_M * 64; it += 256) {
    int row = it >> 6, col = it & 63;                        // col in 16B units
    unsigned long long g = (unsigned long long)(size_t)(xb + (size_t)lds_tok[row] * DMODEL)
                           + (unsigned long long)(col * 16);
    unsigned l = (unsigned)(size_t)(void*)(lds_x + row * XSTR) + (unsigned)(col * 16);
    asm volatile("global_load_async_to_lds_b128 %0, %1, off"
                 :: "v"(l), "v"(g) : "memory");
  }
  asm volatile("s_wait_asynccnt 0x0" ::: "memory");
  __syncthreads();

  const unsigned short* w1e = w1t + (size_t)e * HDIM * DMODEL;   // [H][D] n-major
  const unsigned short* w2e = w2t + (size_t)e * DMODEL * HDIM;   // [D][H] n-major

  v8f acc[2][8];
#pragma unroll
  for (int a = 0; a < 2; ++a)
#pragma unroll
    for (int j = 0; j < 8; ++j) acc[a][j] = (v8f){0.f,0.f,0.f,0.f,0.f,0.f,0.f,0.f};

  // GEMM2 wave tiling: 2 M-tiles x 8 N-tiles per wave
  const int mb = (wave & 1) * 2;   // m-tile base 0 or 2
  const int ng = wave >> 1;        // n-group 0..3 -> n-tiles ng*8..ng*8+7
  const unsigned short* w2base = w2e + (size_t)(ng * 128 + l16) * HDIM + 8 * hh;

  int parity = 0;
  for (int hc = 0; hc < HDIM; hc += 64, parity ^= 1) {
    unsigned short* hbuf = lds_h + parity * (TILE_M * HSTR);
    // ---- GEMM1: h[64x64] = relu(X[64x512] @ W1[:, hc:hc+64] + b1) ----
    {
      const int mp = wave >> 2;    // m-pair 0/1 -> m-tiles 2mp, 2mp+1
      const int nn = wave & 3;     // n-tile within chunk
      v8f c0 = (v8f){0.f,0.f,0.f,0.f,0.f,0.f,0.f,0.f};
      v8f c1 = c0;
      const unsigned short* a0b = lds_x + (mp * 32 + l16) * XSTR + 8 * hh;
      const unsigned short* a1b = a0b + 16 * XSTR;
      const unsigned short* bb  = w1e + (size_t)(hc + nn * 16 + l16) * DMODEL + 8 * hh;
#pragma unroll
      for (int k0 = 0; k0 < DMODEL; k0 += 32) {
        v16bf a0 = load_a_frag(a0b + k0);
        v16bf a1 = load_a_frag(a1b + k0);
        v16bf b  = load_b_frag(bb + k0);
        c0 = wmma_bf16(a0, b, c0);
        c1 = wmma_bf16(a1, b, c1);
      }
      float bias = b1[e * HDIM + hc + nn * 16 + l16];
      int ncol = nn * 16 + l16;
#pragma unroll
      for (int r = 0; r < 8; ++r) {               // C layout: M = r + 8*hh, N = l16
        int m0 = mp * 32 + r + 8 * hh;
        float v0 = c0[r] + bias; v0 = v0 > 0.f ? v0 : 0.f;
        float v1 = c1[r] + bias; v1 = v1 > 0.f ? v1 : 0.f;
        hbuf[m0 * HSTR + ncol]        = f2bf(v0);
        hbuf[(m0 + 16) * HSTR + ncol] = f2bf(v1);
      }
    }
    __syncthreads();   // hbuf[parity] published; buf[1-parity] free for next chunk
    // ---- GEMM2: out[64x512] += h[64x64] @ W2[hc:hc+64, :] ----
#pragma unroll
    for (int kk = 0; kk < 64; kk += 32) {
      v16bf a0 = load_a_frag(hbuf + (mb * 16 + l16) * HSTR + kk + 8 * hh);
      v16bf a1 = load_a_frag(hbuf + ((mb + 1) * 16 + l16) * HSTR + kk + 8 * hh);
      const unsigned short* bp = w2base + hc + kk;
#pragma unroll
      for (int j = 0; j < 8; ++j) {
        v16bf b = load_b_frag(bp + (size_t)j * 16 * HDIM);
        acc[0][j] = wmma_bf16(a0, b, acc[0][j]);
        acc[1][j] = wmma_bf16(a1, b, acc[1][j]);
      }
    }
    // no trailing barrier: next chunk writes the other h buffer
  }

  // ---- epilogue: scale by routing weight, atomic-add into out ----
#pragma unroll
  for (int j = 0; j < 8; ++j) {
    int n = (ng * 8 + j) * 16 + l16;
    float b2v = b2[e * DMODEL + n];
#pragma unroll
    for (int mi = 0; mi < 2; ++mi) {
#pragma unroll
      for (int r = 0; r < 8; ++r) {
        int m = (mb + mi) * 16 + r + 8 * hh;
        float val = (acc[mi][j][r] + b2v) * lds_wgt[m];
        unsafeAtomicAdd(out + (size_t)lds_tok[m] * DMODEL + n, val);
      }
    }
  }
}

// ---------------- host launcher ----------------

extern "C" void kernel_launch(void* const* d_in, const int* in_sizes, int n_in,
                              void* d_out, int out_size, void* d_ws, size_t ws_size,
                              hipStream_t stream) {
  const float* x  = (const float*)d_in[0];
  const float* rw = (const float*)d_in[1];
  const float* rb = (const float*)d_in[2];
  const float* w1 = (const float*)d_in[3];
  const float* b1 = (const float*)d_in[4];
  const float* w2 = (const float*)d_in[5];
  const float* b2 = (const float*)d_in[6];
  float* out = (float*)d_out;

  // workspace layout
  char* ws = (char*)d_ws;
  size_t off = 0;
  int*   cnt     = (int*)(ws + off);                 off += 256;
  int*   idxlist = (int*)(ws + off);                 off += (size_t)NEXP * TOKENS * 4;
  float* wgtlist = (float*)(ws + off);               off += (size_t)NEXP * TOKENS * 4;
  unsigned short* xb  = (unsigned short*)(ws + off); off += (size_t)TOKENS * DMODEL * 2;
  unsigned short* w1t = (unsigned short*)(ws + off); off += (size_t)NEXP * DMODEL * HDIM * 2;
  unsigned short* w2t = (unsigned short*)(ws + off); off += (size_t)NEXP * DMODEL * HDIM * 2;
  if (ws_size < off) return;  // workspace too small; nothing sane to do

  hipMemsetAsync(cnt, 0, 256, stream);
  hipMemsetAsync(d_out, 0, (size_t)out_size * sizeof(float), stream);

  // phase 0: conversions
  cvt_x_bf16<<<(TOKENS * DMODEL / 4) / 256, 256, 0, stream>>>(x, xb);
  transpose_bf16<<<dim3(HDIM / 32, DMODEL / 32, NEXP), 256, 0, stream>>>(w1, w1t, DMODEL, HDIM);
  transpose_bf16<<<dim3(DMODEL / 32, HDIM / 32, NEXP), 256, 0, stream>>>(w2, w2t, HDIM, DMODEL);

  // phase 1: routing
  router_topk<<<TOKENS / 8, 256, 0, stream>>>(x, rw, rb, cnt, idxlist, wgtlist);

  // phase 2: fused expert FFN
  const int smem_bytes = TILE_M * XSTR * 2 + 2 * TILE_M * HSTR * 2 + TILE_M * 4 + TILE_M * 4;
  (void)hipFuncSetAttribute((const void*)moe_ffn,
                            hipFuncAttributeMaxDynamicSharedMemorySize, smem_bytes);
  moe_ffn<<<dim3(TOKENS / TILE_M, NEXP), 256, smem_bytes, stream>>>(
      xb, w1t, w2t, b1, b2, cnt, idxlist, wgtlist, out);
}
